// SSMLayer_38070590112239
// MI455X (gfx1250) — matrix operand order repaired and checked by reading, not compiled, not run on hardware
//
#include <hip/hip_runtime.h>
#include <hip/hip_bf16.h>

// ---------------- problem constants ----------------
#define B_SZ     2
#define T_LEN    1024
#define D_MODEL  1024
#define D_INNER  2048
#define D_STATE  16
#define D_CONV   4
#define DT_RANK  64
#define M_ROWS   (B_SZ * T_LEN)          // 2048 token rows

typedef __attribute__((ext_vector_type(16))) __bf16 v16bf;
typedef __attribute__((ext_vector_type(8)))  float  v8f;

union Frag { v16bf v; uint4 q[2]; };

// LDS tile row pitch: 32 bf16 of data padded to 40 (80 B = 20 banks, avoids
// 64-bank aliasing for the strided ds_load_b128 fragment reads)
#define LPITCH   40
#define LTILE_B  (64 * LPITCH * 2)       // bytes per 64x32 bf16 tile buffer

// ---------------------------------------------------------------------------
// Async-LDS double-buffered bf16 WMMA GEMM: C[M,N] = A[M,K] * W[N,K]^T.
// 128 threads = 4 waves; block computes a 64x64 macro-tile (2x2 wave grid,
// 32x32 per wave). Per K-step (32), A 64x32 and B 64x32 slabs are pulled into
// LDS with global_load_async_to_lds_b128 (ASYNCcnt), double buffered so the
// fill of step i+1 overlaps the 4 WMMAs of step i.
// Requires M%64==0, N%64==0, K%32==0.
// ---------------------------------------------------------------------------
__global__ void __launch_bounds__(128)
gemm_bf16_wmma_lds(const __bf16* __restrict__ A,  // M x K
                   const __bf16* __restrict__ W,  // N x K
                   float* __restrict__ C,         // M x N
                   int M, int N, int K)
{
    __shared__ __align__(16) __bf16 lsA[2][64 * LPITCH];
    __shared__ __align__(16) __bf16 lsB[2][64 * LPITCH];

    const int tid  = threadIdx.x;
    const int wave = tid >> 5;
    const int lane = tid & 31;
    const int half = lane >> 4;          // 0: lanes 0-15, 1: lanes 16-31
    const int l16  = lane & 15;
    const int wm   = wave >> 1;          // wave row in 2x2 grid
    const int wn   = wave & 1;           // wave col

    const int tilesN = N >> 6;
    const int bm = blockIdx.x / tilesN;
    const int bn = blockIdx.x - bm * tilesN;

    // flat shared-aperture address truncates to the LDS byte address (ISA 10.2)
    const uint32_t ldsA0 = (uint32_t)(size_t)(&lsA[0][0]);
    const uint32_t ldsB0 = (uint32_t)(size_t)(&lsB[0][0]);

    // loader: each thread owns 32 contiguous bytes of one row of each slab
    const int ldRow  = tid >> 1;                 // 0..63
    const int ldBoff = (tid & 1) * 32;           // 0 or 32 bytes within 64B k-slab

    const uint8_t* gArow = (const uint8_t*)(A + (size_t)(bm * 64 + ldRow) * K) + ldBoff;
    const uint8_t* gBrow = (const uint8_t*)(W + (size_t)(bn * 64 + ldRow) * K) + ldBoff;
    const uint32_t laApitch = ldRow * (LPITCH * 2) + ldBoff;

    auto issue = [&](int buf, int k0) {
        const uint64_t a64 = (uint64_t)(uintptr_t)(gArow + (size_t)k0 * 2);
        const uint64_t b64 = (uint64_t)(uintptr_t)(gBrow + (size_t)k0 * 2);
        const uint32_t laA = ldsA0 + (uint32_t)buf * LTILE_B + laApitch;
        const uint32_t laB = ldsB0 + (uint32_t)buf * LTILE_B + laApitch;
        // INST_OFFSET is added to both the global and LDS addresses (ISA 15.18)
        asm volatile("global_load_async_to_lds_b128 %0, %1, off"
                     :: "v"(laA), "v"(a64) : "memory");
        asm volatile("global_load_async_to_lds_b128 %0, %1, off offset:16"
                     :: "v"(laA), "v"(a64) : "memory");
        asm volatile("global_load_async_to_lds_b128 %0, %1, off"
                     :: "v"(laB), "v"(b64) : "memory");
        asm volatile("global_load_async_to_lds_b128 %0, %1, off offset:16"
                     :: "v"(laB), "v"(b64) : "memory");
    };

    v8f acc[2][2] = {};
    const int nk = K >> 5;

    issue(0, 0);
    for (int i = 0; i < nk; ++i) {
        const int buf = i & 1;
        asm volatile("s_wait_asynccnt 0x0" ::: "memory");
        __syncthreads();                          // buf is visible to all waves
        if (i + 1 < nk) issue(buf ^ 1, (i + 1) << 5);   // overlap next fill

        const __bf16* sA = &lsA[buf][0];
        const __bf16* sB = &lsB[buf][0];
        Frag a[2], b[2];
#pragma unroll
        for (int mm = 0; mm < 2; ++mm) {
            // A frag: lane<16 -> row l16, K {0..7,16..23}; lane>=16 -> K {8..15,24..31}
            const __bf16* p = sA + (wm * 32 + mm * 16 + l16) * LPITCH + (half ? 8 : 0);
            a[mm].q[0] = *(const uint4*)(p);
            a[mm].q[1] = *(const uint4*)(p + 16);
        }
#pragma unroll
        for (int nn = 0; nn < 2; ++nn) {
            // B frag: lane<16 -> col l16, K {0..7,16..23}; lane>=16 -> K {8..15,24..31}
            const __bf16* p = sB + (wn * 32 + nn * 16 + l16) * LPITCH + (half ? 8 : 0);
            b[nn].q[0] = *(const uint4*)(p);
            b[nn].q[1] = *(const uint4*)(p + 16);
        }
#pragma unroll
        for (int mm = 0; mm < 2; ++mm)
#pragma unroll
            for (int nn = 0; nn < 2; ++nn)
                acc[mm][nn] = __builtin_amdgcn_wmma_f32_16x16x32_bf16(
                    false, a[mm].v, false, b[nn].v, (short)0, acc[mm][nn], false, false);
        __syncthreads();                          // done reading buf before refill
    }

    // C/D layout: VGPR v, lane l -> M = v + 8*(l>=16), N = l%16
#pragma unroll
    for (int mm = 0; mm < 2; ++mm)
#pragma unroll
        for (int nn = 0; nn < 2; ++nn) {
            const int n = bn * 64 + wn * 32 + nn * 16 + l16;
#pragma unroll
            for (int v = 0; v < 8; ++v) {
                const int m = bm * 64 + wm * 32 + mm * 16 + v + half * 8;
                C[(size_t)m * N + n] = acc[mm][nn][v];
            }
        }
}

// ---------------------------------------------------------------------------
// Direct-load WMMA GEMM (used for N%64 != 0, i.e. the 96-wide x_proj GEMM).
// One wave per 32x32 tile, fragments loaded straight from global (hits L2).
// ---------------------------------------------------------------------------
__global__ void __launch_bounds__(128)
gemm_bf16_wmma(const __bf16* __restrict__ A,  // M x K
               const __bf16* __restrict__ W,  // N x K
               float* __restrict__ C,         // M x N
               int M, int N, int K)
{
    const int wave   = threadIdx.x >> 5;
    const int lane   = threadIdx.x & 31;
    const int tilesN = N >> 5;
    const int tileId = blockIdx.x * (blockDim.x >> 5) + wave;
    const int tm     = tileId / tilesN;
    const int tn     = tileId - tm * tilesN;
    if (tm * 32 >= M) return;

    const int half = lane >> 4;
    const int l16  = lane & 15;

    v8f acc[2][2] = {};
    const int aRow0 = tm * 32 + l16;
    const int bCol0 = tn * 32 + l16;

    for (int k0 = 0; k0 < K; k0 += 32) {
        Frag a[2], b[2];
#pragma unroll
        for (int mm = 0; mm < 2; ++mm) {
            const __bf16* p = A + (size_t)(aRow0 + mm * 16) * K + k0 + (half ? 8 : 0);
            a[mm].q[0] = *(const uint4*)(p);
            a[mm].q[1] = *(const uint4*)(p + 16);
            __builtin_prefetch((const void*)(p + 32), 0, 1);   // speculative: safe OOB
        }
#pragma unroll
        for (int nn = 0; nn < 2; ++nn) {
            const __bf16* p = W + (size_t)(bCol0 + nn * 16) * K + k0 + (half ? 8 : 0);
            b[nn].q[0] = *(const uint4*)(p);
            b[nn].q[1] = *(const uint4*)(p + 16);
            __builtin_prefetch((const void*)(p + 32), 0, 1);
        }
#pragma unroll
        for (int mm = 0; mm < 2; ++mm)
#pragma unroll
            for (int nn = 0; nn < 2; ++nn)
                acc[mm][nn] = __builtin_amdgcn_wmma_f32_16x16x32_bf16(
                    false, a[mm].v, false, b[nn].v, (short)0, acc[mm][nn], false, false);
    }

#pragma unroll
    for (int mm = 0; mm < 2; ++mm)
#pragma unroll
        for (int nn = 0; nn < 2; ++nn) {
            const int n = tn * 32 + nn * 16 + l16;
#pragma unroll
            for (int v = 0; v < 8; ++v) {
                const int m = tm * 32 + mm * 16 + v + half * 8;
                C[(size_t)m * N + n] = acc[mm][nn][v];
            }
        }
}

// ---------------- fp32 -> bf16 convert ----------------
__global__ void f32_to_bf16(const float* __restrict__ in, __bf16* __restrict__ out, int n)
{
    int i = blockIdx.x * blockDim.x + threadIdx.x;
    if (i < n) out[i] = (__bf16)in[i];
}

// ---------------- causal depthwise conv1d + SiLU ----------------
__global__ void conv_silu(const float* __restrict__ xz, const float* __restrict__ cw,
                          const float* __restrict__ cb, float* __restrict__ xs_f,
                          __bf16* __restrict__ xs_b)
{
    int idx = blockIdx.x * blockDim.x + threadIdx.x;
    if (idx >= M_ROWS * D_INNER) return;
    const int d  = idx % D_INNER;
    const int bt = idx / D_INNER;
    const int t  = bt % T_LEN;
    const int b  = bt / T_LEN;

    float acc = cb[d];
#pragma unroll
    for (int j = 0; j < D_CONV; ++j) {
        const int tt = t - (D_CONV - 1) + j;
        if (tt >= 0)
            acc += cw[d * D_CONV + j] * xz[((size_t)(b * T_LEN + tt)) * (2 * D_INNER) + d];
    }
    const float s = acc / (1.0f + __expf(-acc));   // silu
    xs_f[idx] = s;
    xs_b[idx] = (__bf16)s;
}

// ---------------- slice dt-rank columns of x_dbl -> bf16 ----------------
__global__ void extract_dt(const float* __restrict__ xdbl, __bf16* __restrict__ dtin)
{
    int i = blockIdx.x * blockDim.x + threadIdx.x;
    if (i >= M_ROWS * DT_RANK) return;
    const int r = i / DT_RANK, c = i - r * DT_RANK;
    dtin[i] = (__bf16)xdbl[(size_t)r * (DT_RANK + 2 * D_STATE) + c];
}

// ---------------- softplus(dt + bias) ----------------
__global__ void softplus_bias(float* __restrict__ dt, const float* __restrict__ bias)
{
    int i = blockIdx.x * blockDim.x + threadIdx.x;
    if (i >= M_ROWS * D_INNER) return;
    const float v = dt[i] + bias[i % D_INNER];
    dt[i] = (v > 20.0f) ? v : log1pf(__expf(v));
}

// ---------------- selective scan ----------------
// one thread per (b, d, s); 16-lane butterfly reduce for y_t = <h_t, C_t>
__global__ void selective_scan(const float* __restrict__ dt,
                               const float* __restrict__ xdbl,
                               const float* __restrict__ xs,
                               const float* __restrict__ A_log,
                               const float* __restrict__ Dp,
                               float* __restrict__ y)
{
    const int gid = blockIdx.x * blockDim.x + threadIdx.x;
    const int s  = gid & (D_STATE - 1);
    const int ch = gid >> 4;
    const int d  = ch % D_INNER;
    const int b  = ch / D_INNER;

    const float a  = -__expf(A_log[d * D_STATE + s]);
    const float Dv = Dp[d];
    float h = 0.0f;

    const float* dt_p = dt   + (size_t)b * T_LEN * D_INNER + d;
    const float* xs_p = xs   + (size_t)b * T_LEN * D_INNER + d;
    const float* bc_p = xdbl + (size_t)b * T_LEN * (DT_RANK + 2 * D_STATE);
    float*       y_p  = y    + (size_t)b * T_LEN * D_INNER + d;

    for (int t = 0; t < T_LEN; ++t) {
        const float dtv = dt_p[(size_t)t * D_INNER];
        const float xv  = xs_p[(size_t)t * D_INNER];
        const float Bv  = bc_p[t * (DT_RANK + 2 * D_STATE) + DT_RANK + s];
        const float Cv  = bc_p[t * (DT_RANK + 2 * D_STATE) + DT_RANK + D_STATE + s];

        const float dA = __expf(dtv * a);
        h = dA * h + (dtv * xv) * Bv;

        float p = h * Cv;
        p += __shfl_xor(p, 1, 16);
        p += __shfl_xor(p, 2, 16);
        p += __shfl_xor(p, 4, 16);
        p += __shfl_xor(p, 8, 16);
        if (s == 0) y_p[(size_t)t * D_INNER] = p + xv * Dv;
    }
}

// ---------------- gate: y * silu(z) -> bf16 ----------------
__global__ void gate_silu(const float* __restrict__ y, const float* __restrict__ xz,
                          __bf16* __restrict__ yb)
{
    int idx = blockIdx.x * blockDim.x + threadIdx.x;
    if (idx >= M_ROWS * D_INNER) return;
    const int row = idx / D_INNER;
    const int d   = idx - row * D_INNER;
    const float z = xz[(size_t)row * (2 * D_INNER) + D_INNER + d];
    const float g = z / (1.0f + __expf(-z));
    yb[idx] = (__bf16)(y[idx] * g);
}

// ---------------------------------------------------------------------------
static inline int cdiv(int a, int b) { return (a + b - 1) / b; }

extern "C" void kernel_launch(void* const* d_in, const int* in_sizes, int n_in,
                              void* d_out, int out_size, void* d_ws, size_t ws_size,
                              hipStream_t stream)
{
    const float* x         = (const float*)d_in[0];
    const float* in_proj_w = (const float*)d_in[1];
    const float* conv_w    = (const float*)d_in[2];
    const float* conv_b    = (const float*)d_in[3];
    const float* x_proj_w  = (const float*)d_in[4];
    const float* dt_proj_w = (const float*)d_in[5];
    const float* dt_proj_b = (const float*)d_in[6];
    const float* A_log     = (const float*)d_in[7];
    const float* D_param   = (const float*)d_in[8];
    const float* out_proj_w= (const float*)d_in[9];
    float* out = (float*)d_out;

    // ------- workspace carve-out (all offsets 256B aligned) -------
    char* ws = (char*)d_ws;
    size_t o = 0;
    auto carve = [&](size_t bytes) -> void* {
        void* p = ws + o;
        o = (o + bytes + 255) & ~(size_t)255;
        return p;
    };
    __bf16* x_bf      = (__bf16*)carve((size_t)M_ROWS * D_MODEL * 2);
    __bf16* w_in_bf   = (__bf16*)carve((size_t)2 * D_INNER * D_MODEL * 2);
    float*  xz        = (float*) carve((size_t)M_ROWS * 2 * D_INNER * 4);
    float*  xs_f      = (float*) carve((size_t)M_ROWS * D_INNER * 4);
    __bf16* xs_bf     = (__bf16*)carve((size_t)M_ROWS * D_INNER * 2);
    __bf16* w_xp_bf   = (__bf16*)carve((size_t)(DT_RANK + 2 * D_STATE) * D_INNER * 2);
    float*  xdbl      = (float*) carve((size_t)M_ROWS * (DT_RANK + 2 * D_STATE) * 4);
    __bf16* dtin_bf   = (__bf16*)carve((size_t)M_ROWS * DT_RANK * 2);
    __bf16* w_dt_bf   = (__bf16*)carve((size_t)D_INNER * DT_RANK * 2);
    float*  dt_f      = (float*) carve((size_t)M_ROWS * D_INNER * 4);
    float*  y_f       = (float*) carve((size_t)M_ROWS * D_INNER * 4);
    __bf16* y_bf      = (__bf16*)carve((size_t)M_ROWS * D_INNER * 2);
    __bf16* w_out_bf  = (__bf16*)carve((size_t)D_MODEL * D_INNER * 2);
    (void)ws_size; (void)in_sizes; (void)n_in; (void)out_size;

    const int TB = 256;

    // ------- stage 0: bf16 conversions (weights + input) -------
    f32_to_bf16<<<cdiv(M_ROWS * D_MODEL, TB), TB, 0, stream>>>(x, x_bf, M_ROWS * D_MODEL);
    f32_to_bf16<<<cdiv(2 * D_INNER * D_MODEL, TB), TB, 0, stream>>>(in_proj_w, w_in_bf, 2 * D_INNER * D_MODEL);
    f32_to_bf16<<<cdiv((DT_RANK + 2 * D_STATE) * D_INNER, TB), TB, 0, stream>>>(x_proj_w, w_xp_bf, (DT_RANK + 2 * D_STATE) * D_INNER);
    f32_to_bf16<<<cdiv(D_INNER * DT_RANK, TB), TB, 0, stream>>>(dt_proj_w, w_dt_bf, D_INNER * DT_RANK);
    f32_to_bf16<<<cdiv(D_MODEL * D_INNER, TB), TB, 0, stream>>>(out_proj_w, w_out_bf, D_MODEL * D_INNER);

    // ------- GEMM 1: xz = x @ in_proj_w^T   (2048 x 4096, K=1024) -------
    gemm_bf16_wmma_lds<<<(M_ROWS / 64) * ((2 * D_INNER) / 64), 128, 0, stream>>>(
        x_bf, w_in_bf, xz, M_ROWS, 2 * D_INNER, D_MODEL);

    // ------- conv1d + silu -------
    conv_silu<<<cdiv(M_ROWS * D_INNER, TB), TB, 0, stream>>>(xz, conv_w, conv_b, xs_f, xs_bf);

    // ------- GEMM 2: x_dbl = xs @ x_proj_w^T  (2048 x 96, K=2048; N%64!=0) -------
    {
        int tiles = (M_ROWS / 32) * ((DT_RANK + 2 * D_STATE) / 32);
        gemm_bf16_wmma<<<cdiv(tiles, 4), 128, 0, stream>>>(xs_bf, w_xp_bf, xdbl,
                                                           M_ROWS, DT_RANK + 2 * D_STATE, D_INNER);
    }

    // ------- dt slice -> bf16; GEMM 3: dt = dt_in @ dt_proj_w^T (2048 x 2048, K=64) -------
    extract_dt<<<cdiv(M_ROWS * DT_RANK, TB), TB, 0, stream>>>(xdbl, dtin_bf);
    gemm_bf16_wmma_lds<<<(M_ROWS / 64) * (D_INNER / 64), 128, 0, stream>>>(
        dtin_bf, w_dt_bf, dt_f, M_ROWS, D_INNER, DT_RANK);
    softplus_bias<<<cdiv(M_ROWS * D_INNER, TB), TB, 0, stream>>>(dt_f, dt_proj_b);

    // ------- selective scan -------
    selective_scan<<<(B_SZ * D_INNER * D_STATE) / TB, TB, 0, stream>>>(
        dt_f, xdbl, xs_f, A_log, D_param, y_f);

    // ------- gate with silu(z) -------
    gate_silu<<<cdiv(M_ROWS * D_INNER, TB), TB, 0, stream>>>(y_f, xz, y_bf);

    // ------- GEMM 4: out = y @ out_proj_w^T  (2048 x 1024, K=2048) -------
    gemm_bf16_wmma_lds<<<(M_ROWS / 64) * (D_MODEL / 64), 128, 0, stream>>>(
        y_bf, w_out_bf, out, M_ROWS, D_MODEL, D_INNER);
}
